// Model_1331439862409
// MI455X (gfx1250) — matrix operand order repaired and checked by reading, not compiled, not run on hardware
//
#include <hip/hip_runtime.h>
#include <hip/hip_bf16.h>
#include <math.h>

// ---------------- problem constants ----------------
#define BATCH 512
#define SEQ   512
#define EMB   100
#define EMBP  128          // padded emb dim (multiple of 32)
#define HID   512
#define NG    1536         // 3*HID
#define KTOT  640          // EMBP + HID
#define NKT   20           // KTOT/32 k-steps
#define NNT   96           // NG/16 n-tiles
#define NHT   32           // HID/16 n-tiles per gate
#define NMT   32           // BATCH/16 m-tiles

// ---------------- workspace layout (bytes), all 256-aligned ----------------
#define OFF_W     ((size_t)0)                  // NKT*NNT*512 halves = 1,966,080 B
#define OFF_BIAS  ((size_t)1966080)            // 4*HID floats
#define OFF_EMB   ((size_t)1974272)            // SEQ*BATCH*EMBP halves = 64 MB
#define OFF_H0    ((size_t)69083136)           // BATCH*HID halves
#define OFF_H1    ((size_t)69607424)
// total ~70.2 MB

typedef __attribute__((ext_vector_type(16))) _Float16 v16h;
typedef __attribute__((ext_vector_type(8)))  _Float16 v8h;
typedef __attribute__((ext_vector_type(8)))  float    v8f;
typedef __attribute__((ext_vector_type(4)))  unsigned v4u;
typedef __attribute__((ext_vector_type(8)))  int      v8i;
typedef __attribute__((ext_vector_type(4)))  int      v4i;

// ---------------------------------------------------------------------------
// Pack W[k][n] (k<100: w_ih[n,k]; 128<=k<640: w_hh[n,k-128]; else 0) into the
// CDNA5 WMMA B-fragment layout for v_wmma_f32_16x16x32_f16:
//   frag ft = kt*NNT + gnt; col = gnt*16+(lane&15), g = lane>>4;
//                           K = kt*32 + (h<8 ? 8g+h : 16+8g+h-8).
// Stored contiguously: Wp[ft*512 + lane*16 + h]  (32B per lane per frag)
// ---------------------------------------------------------------------------
__global__ __launch_bounds__(256)
void pack_w_kernel(const float* __restrict__ w_ih, const float* __restrict__ w_hh,
                   _Float16* __restrict__ Wp) {
    int t = blockIdx.x * 256 + threadIdx.x;
    if (t >= NKT * NNT * 512) return;
    int h    = t & 15;
    int lane = (t >> 4) & 31;
    int ft   = t >> 9;
    int gnt  = ft % NNT;
    int kt   = ft / NNT;
    int n    = gnt * 16 + (lane & 15);
    int k    = kt * 32 + ((h >> 3) << 4) + ((lane >> 4) << 3) + (h & 7);
    float v = 0.f;
    if (k < EMB)        v = w_ih[(size_t)n * EMB + k];
    else if (k >= EMBP) v = w_hh[(size_t)n * HID + (k - EMBP)];
    Wp[t] = (_Float16)v;
}

// bias4: [0]=b_ih_r+b_hh_r, [1]=b_ih_z+b_hh_z, [2]=b_ih_n, [3]=b_hh_n
__global__ __launch_bounds__(256)
void pack_bias_kernel(const float* __restrict__ b_ih, const float* __restrict__ b_hh,
                      float* __restrict__ bias4) {
    int j = blockIdx.x * 256 + threadIdx.x;
    if (j >= HID) return;
    bias4[0 * HID + j] = b_ih[j]           + b_hh[j];
    bias4[1 * HID + j] = b_ih[HID + j]     + b_hh[HID + j];
    bias4[2 * HID + j] = b_ih[2 * HID + j];
    bias4[3 * HID + j] = b_hh[2 * HID + j];
}

// Gather embeddings -> time-major f16 [SEQ][BATCH][EMBP], zero-padded cols.
__global__ __launch_bounds__(256)
void embed_kernel(const int* __restrict__ x, const float* __restrict__ tab,
                  _Float16* __restrict__ embf) {
    size_t t = (size_t)blockIdx.x * 256 + threadIdx.x;   // < SEQ*BATCH*EMBP
    int    k  = (int)(t & (EMBP - 1));
    size_t sb = t >> 7;                                  // EMBP = 128
    int    b  = (int)(sb & (BATCH - 1));
    int    s  = (int)(sb >> 9);                          // BATCH = 512
    float v = 0.f;
    if (k < EMB) {
        int idx = x[(size_t)b * SEQ + s];
        v = tab[(size_t)idx * EMB + k];
    }
    embf[t] = (_Float16)v;
}

__global__ __launch_bounds__(256)
void zero_h_kernel(_Float16* __restrict__ h) {
    int t = blockIdx.x * 256 + threadIdx.x;
    if (t < BATCH * HID) h[t] = (_Float16)0.f;
}

// ---------------------------------------------------------------------------
// One GRU timestep.  1024 wave-tasks: (mt,nt) in 32x32; the 4 waves of a
// block share one nt, so the block's full B working set (20 kt x 3 gates x
// 1KB = 60KB) is staged once into LDS by a single Tensor Data Mover
// descriptor (3D tile: x=512 halves, y=gate stride 16384, z=kt stride 49152),
// then consumed by all 4 waves via ds_load.  Each wave accumulates 4 gate
// tiles over K=640 (20 k-steps, 3 v_wmma each).
// ---------------------------------------------------------------------------
__global__ __launch_bounds__(128)
void gru_step_kernel(const _Float16* __restrict__ xt,    // [BATCH][EMBP]
                     const _Float16* __restrict__ Wp,    // packed B frags
                     const float*    __restrict__ bias4, // [4][HID]
                     const _Float16* __restrict__ hprev, // [BATCH][HID]
                     _Float16*       __restrict__ hnext) // [BATCH][HID]
{
    __shared__ __align__(32) _Float16 Bsh[NKT * 3 * 512];   // 60 KB

    const int lane = threadIdx.x & 31;
    const int wave = threadIdx.x >> 5;
    const int task = blockIdx.x * 4 + wave;   // 0..1023
    const int mt   = task & 31;
    const int nt   = blockIdx.x >> 3;         // same for all 4 waves in block
    const int ncol = nt * 16 + (lane & 15);
    const int mrow = mt * 16 + (lane & 15);
    const int koff = (lane >> 4) * 8;         // ISA A-layout half-offset

    if (wave == 0) {
        // ---- TDM: load 60KB of B fragments for this nt into LDS ----
        unsigned lds0 = (unsigned)(size_t)(&Bsh[0]);                 // LDS byte offset
        unsigned long long ga = (unsigned long long)(size_t)(Wp + (size_t)nt * 512);
        // D# group 0: count=1 | lds_addr | global_addr[56:0] | type=2
        v4u g0 = { 1u,
                   lds0,
                   (unsigned)(ga & 0xffffffffu),
                   (unsigned)((ga >> 32) & 0x01ffffffu) | 0x80000000u };
        // D# group 1: data_size=2B; tensor_dim0=512, tensor_dim1=3;
        //             tile_dim0=512, tile_dim1=3, tile_dim2=20;
        //             dim0_stride=16384 (gate), dim1_stride=49152 (kt)
        v8i g1 = { 0x00010000,            // mask=0, data_size=1 (2B)
                   0x02000000,            // tensor_dim0[15:0]=512 << 16
                   0x00030000,            // tensor_dim0 hi=0 | tensor_dim1[15:0]=3
                   0x02000000,            // tensor_dim1 hi=0 | tile_dim0=512
                   0x00140003,            // tile_dim1=3 | tile_dim2=20
                   16384,                 // tensor_dim0_stride lo32
                   (int)0xC0000000u,      // d0s hi=0 | tensor_dim1_stride[15:0]=0xC000
                   0 };                   // tensor_dim1_stride hi
        v4i g2 = { 20, 0, 0, 0 };         // tensor_dim2=20; tile_dim3=0
        v4i g3 = { 0, 0, 0, 0 };
        v8i g4 = { 0, 0, 0, 0, 0, 0, 0, 0 };   // extra group (6-arg form), unused
        __builtin_amdgcn_tensor_load_to_lds(g0, g1, g2, g3, g4, 0);
        __builtin_amdgcn_s_wait_tensorcnt(0);
    }
    __syncthreads();

    float bv;
    bv = bias4[0 * HID + ncol]; v8f accr  = {bv, bv, bv, bv, bv, bv, bv, bv};
    bv = bias4[1 * HID + ncol]; v8f accz  = {bv, bv, bv, bv, bv, bv, bv, bv};
    bv = bias4[2 * HID + ncol]; v8f accxn = {bv, bv, bv, bv, bv, bv, bv, bv};
    bv = bias4[3 * HID + ncol]; v8f acchn = {bv, bv, bv, bv, bv, bv, bv, bv};

    const _Float16* arow_x = xt    + (size_t)mrow * EMBP;
    const _Float16* arow_h = hprev + (size_t)mrow * HID;

    for (int kt = 0; kt < NKT; ++kt) {
        const _Float16* ap = (kt < 4) ? (arow_x + kt * 32)
                                      : (arow_h + (kt - 4) * 32);
        union { v16h v; v8h p[2]; } ua;
        ua.p[0] = *(const v8h*)(ap + koff);        // K koff..koff+7
        ua.p[1] = *(const v8h*)(ap + 16 + koff);   // K 16+koff..16+koff+7
        v16h a = ua.v;

        const _Float16* bb = &Bsh[(size_t)(kt * 3) * 512 + lane * 16];
        v16h b_r = *(const v16h*)(bb + 0 * 512);
        v16h b_z = *(const v16h*)(bb + 1 * 512);
        v16h b_n = *(const v16h*)(bb + 2 * 512);

        accr = __builtin_amdgcn_wmma_f32_16x16x32_f16(false, a, false, b_r,
                                                      (short)0, accr, false, false);
        accz = __builtin_amdgcn_wmma_f32_16x16x32_f16(false, a, false, b_z,
                                                      (short)0, accz, false, false);
        if (kt < 4)
            accxn = __builtin_amdgcn_wmma_f32_16x16x32_f16(false, a, false, b_n,
                                                           (short)0, accxn, false, false);
        else
            acchn = __builtin_amdgcn_wmma_f32_16x16x32_f16(false, a, false, b_n,
                                                           (short)0, acchn, false, false);
    }

    // C/D layout: element r of acc -> (M = mt*16 + r + (lane>=16 ? 8:0), N = ncol)
    const int mbase = mt * 16 + ((lane >> 4) ? 8 : 0);
#pragma unroll
    for (int r = 0; r < 8; ++r) {
        float rg = 1.f / (1.f + __expf(-accr[r]));
        float zg = 1.f / (1.f + __expf(-accz[r]));
        float ng = tanhf(accxn[r] + rg * acchn[r]);
        float hv = (float)hprev[(size_t)(mbase + r) * HID + ncol];
        float hn = (1.f - zg) * ng + zg * hv;
        hnext[(size_t)(mbase + r) * HID + ncol] = (_Float16)hn;
    }
}

// out[b][o] = h[b] . fc_w[o] + fc_b[o]   (512 x 2, trivial)
__global__ __launch_bounds__(256)
void fc_kernel(const _Float16* __restrict__ h, const float* __restrict__ fc_w,
               const float* __restrict__ fc_b, float* __restrict__ out) {
    int t = blockIdx.x * 256 + threadIdx.x;
    if (t >= BATCH * 2) return;
    int b = t >> 1, o = t & 1;
    float s = fc_b[o];
    const _Float16* hp = h + (size_t)b * HID;
    const float*    wp = fc_w + (size_t)o * HID;
    for (int j = 0; j < HID; ++j) s += (float)hp[j] * wp[j];
    out[t] = s;
}

extern "C" void kernel_launch(void* const* d_in, const int* in_sizes, int n_in,
                              void* d_out, int out_size, void* d_ws, size_t ws_size,
                              hipStream_t stream) {
    const int*   x     = (const int*)  d_in[0];
    const float* tab   = (const float*)d_in[1];
    const float* w_ih  = (const float*)d_in[2];
    const float* w_hh  = (const float*)d_in[3];
    const float* b_ih  = (const float*)d_in[4];
    const float* b_hh  = (const float*)d_in[5];
    const float* fc_w  = (const float*)d_in[6];
    const float* fc_b  = (const float*)d_in[7];
    float*       out   = (float*)d_out;

    char* ws = (char*)d_ws;
    _Float16* Wp    = (_Float16*)(ws + OFF_W);
    float*    bias4 = (float*)   (ws + OFF_BIAS);
    _Float16* embf  = (_Float16*)(ws + OFF_EMB);
    _Float16* hA    = (_Float16*)(ws + OFF_H0);
    _Float16* hB    = (_Float16*)(ws + OFF_H1);

    pack_w_kernel  <<<(NKT * NNT * 512 + 255) / 256, 256, 0, stream>>>(w_ih, w_hh, Wp);
    pack_bias_kernel<<<2, 256, 0, stream>>>(b_ih, b_hh, bias4);
    embed_kernel   <<<(int)(((size_t)SEQ * BATCH * EMBP) / 256), 256, 0, stream>>>(x, tab, embf);
    zero_h_kernel  <<<(BATCH * HID + 255) / 256, 256, 0, stream>>>(hA);

    const _Float16* hprev = hA;
    _Float16*       hnext = hB;
    for (int s = 0; s < SEQ; ++s) {
        gru_step_kernel<<<256, 128, 0, stream>>>(embf + (size_t)s * BATCH * EMBP,
                                                 Wp, bias4, hprev, hnext);
        const _Float16* t = hprev; hprev = hnext; hnext = (_Float16*)t;
    }
    fc_kernel<<<(BATCH * 2 + 255) / 256, 256, 0, stream>>>(hprev, fc_w, fc_b, out);
}
